// CDFA_30863634989618
// MI455X (gfx1250) — matrix operand order repaired and checked by reading, not compiled, not run on hardware
//
#include <hip/hip_runtime.h>
#include <hip/hip_bf16.h>

// ---------------------------------------------------------------------------
// CDNA5 (gfx1250) wave32 WMMA implementation of the CDFA reference pipeline.
// All GEMM-shaped work (4 convs as implicit GEMM, 5 linears) runs on
// v_wmma_f32_16x16x32_bf16; softmax + attention-apply/fold stay in VALU f32.
// Register blocking: each wave owns 2 M-tiles (32 pixels) x 1 N-tile, sharing
// one B fragment across two WMMAs; inner K loops fully unrolled for
// load/WMMA overlap.
// ---------------------------------------------------------------------------

typedef __attribute__((ext_vector_type(16))) __bf16        v16bf;
typedef __attribute__((ext_vector_type(8)))  float         v8f;
typedef __attribute__((ext_vector_type(4)))  unsigned int  u32x4;

union BF16Frag { v16bf v; u32x4 q[2]; };

#define HH    128
#define WW    128
#define NB    2
#define HPAD  130
#define WPAD  130
#define NPIX  32768          // NB*HH*WW
#define MBLK  1024           // NPIX/32  (2 M-tiles per block-wave)
#define ATT_STRIDE 336       // 324 padded to 21 N-tiles
#define ATT_N      324
#define ATT_SCALE  0.17677669529663687f   // 32^-0.5
#define BN_EPS     1e-5f

__device__ __forceinline__ unsigned short f2bf(float f) {
  unsigned u = __float_as_uint(f);
  unsigned r = 0x7FFFu + ((u >> 16) & 1u);       // round-to-nearest-even
  return (unsigned short)((u + r) >> 16);
}
__device__ __forceinline__ float bf2f(unsigned short s) {
  return __uint_as_float(((unsigned)s) << 16);
}

__device__ __forceinline__ v8f wmma_bf16(const BF16Frag& a, const BF16Frag& b,
                                         v8f c) {
  return __builtin_amdgcn_wmma_f32_16x16x32_bf16(
      false, a.v, false, b.v, (short)0, c, false, false);
}

// ---------------------------------------------------------------------------
// utility kernels
// ---------------------------------------------------------------------------
__global__ void k_fill0(u32x4* __restrict__ p, long n) {
  long i = (long)blockIdx.x * blockDim.x + threadIdx.x;
  if (i < n) { u32x4 z = {0u, 0u, 0u, 0u}; p[i] = z; }
}

// f32 NCHW -> bf16 padded NHWC [b][h+1][w+1][c]
__global__ void k_nchw_to_padbf16(const float* __restrict__ in,
                                  unsigned short* __restrict__ out, int C) {
  long t = (long)blockIdx.x * blockDim.x + threadIdx.x;
  long total = (long)NB * C * HH * WW;
  if (t >= total) return;
  int w = (int)(t % WW);
  int h = (int)((t / WW) % HH);
  int c = (int)((t / ((long)WW * HH)) % C);
  int b = (int)(t / ((long)WW * HH * C));
  out[((long)(b * HPAD + h + 1) * WPAD + (w + 1)) * C + c] = f2bf(in[t]);
}

// f32 NCHW -> bf16 flat NHWC [pix][c]
__global__ void k_nchw_to_flatbf16(const float* __restrict__ in,
                                   unsigned short* __restrict__ out, int C) {
  long t = (long)blockIdx.x * blockDim.x + threadIdx.x;
  long total = (long)NB * C * HH * WW;
  if (t >= total) return;
  int w = (int)(t % WW);
  int h = (int)((t / WW) % HH);
  int c = (int)((t / ((long)WW * HH)) % C);
  int b = (int)(t / ((long)WW * HH * C));
  out[((long)((b * HH + h) * WW + w)) * C + c] = f2bf(in[t]);
}

// conv weight OIHW f32 -> bf16 [co][tap*Ci + ci], tap = kh*3+kw
__global__ void k_pack_convw(const float* __restrict__ w,
                             unsigned short* __restrict__ out, int Co, int Ci) {
  long t = (long)blockIdx.x * blockDim.x + threadIdx.x;
  long total = (long)Co * 9 * Ci;
  if (t >= total) return;
  int ci  = (int)(t % Ci);
  int tap = (int)((t / Ci) % 9);
  int co  = (int)(t / ((long)Ci * 9));
  int kh = tap / 3, kw = tap % 3;
  out[t] = f2bf(w[(((long)co * Ci + ci) * 3 + kh) * 3 + kw]);
}

// f32 [N][Kc] -> bf16 [Npad][Kc], zero padded rows
__global__ void k_pack_mat(const float* __restrict__ w,
                           unsigned short* __restrict__ out,
                           int N, int Kc, int Npad) {
  long t = (long)blockIdx.x * blockDim.x + threadIdx.x;
  long total = (long)Npad * Kc;
  if (t >= total) return;
  int n = (int)(t / Kc);
  int k = (int)(t % Kc);
  out[t] = n < N ? f2bf(w[(long)n * Kc + k]) : (unsigned short)0;
}

// ---------------------------------------------------------------------------
// implicit-GEMM conv3x3 + BN + ReLU.  8 waves/block, wave = N-tile (Cout=128).
// Each wave: 2 M-tiles (32 pixels) sharing one B fragment per K chunk.
// mode 0: bf16 padded NHWC out; mode 1: bf16 flat NHWC; mode 2: f32 NCHW.
// ---------------------------------------------------------------------------
template <int CIN>
__global__ __launch_bounds__(256) void conv_gemm(
    const unsigned short* __restrict__ xpad,   // [NB][130][130][CIN] bf16
    const unsigned short* __restrict__ wp,     // [128][9*CIN] bf16
    const float* __restrict__ gmm, const float* __restrict__ bet,
    const float* __restrict__ mu,  const float* __restrict__ var,
    unsigned short* __restrict__ outb, float* __restrict__ outf, int mode) {
  const int lane = threadIdx.x & 31;
  const int wv   = threadIdx.x >> 5;     // 0..7 -> N tile
  const int hi   = lane >> 4;
  const int lm   = lane & 15;
  const int pix0 = blockIdx.x << 5;      // 32 consecutive pixels (same row)
  const int b    = pix0 >> 14;
  const int h    = (pix0 >> 7) & 127;
  const int w0   = pix0 & 127;

  constexpr int KT = CIN * 9;
  const long abase = ((long)(b * HPAD + h) * WPAD + (w0 + lm)) * CIN + (hi << 3);
  const unsigned short* bp = wp + (long)((wv << 4) + lm) * KT + (hi << 3);

  v8f acc[2] = {{}, {}};
  for (int tap = 0; tap < 9; ++tap) {
    const int ki = tap / 3, kj = tap - 3 * ki;       // strength-reduced by LLVM
    const long toff = (long)(ki * WPAD + kj) * CIN;
    const unsigned short* ap0 = xpad + abase + toff;
    const unsigned short* bq0 = bp + tap * CIN;
#pragma unroll
    for (int cb = 0; cb < CIN; cb += 32) {
      BF16Frag a0, a1, bb;
      const unsigned short* ap = ap0 + cb;
      a0.q[0] = *(const u32x4*)(ap);
      a0.q[1] = *(const u32x4*)(ap + 16);
      a1.q[0] = *(const u32x4*)(ap + 16 * CIN);
      a1.q[1] = *(const u32x4*)(ap + 16 * CIN + 16);
      const unsigned short* bq = bq0 + cb;
      bb.q[0] = *(const u32x4*)(bq);
      bb.q[1] = *(const u32x4*)(bq + 16);
      acc[0] = wmma_bf16(a0, bb, acc[0]);
      acc[1] = wmma_bf16(a1, bb, acc[1]);
    }
  }

  const int co = (wv << 4) + lm;
  const float sc = gmm[co] * rsqrtf(var[co] + BN_EPS);
  const float sh = bet[co] - mu[co] * sc;
#pragma unroll
  for (int t = 0; t < 2; ++t) {
#pragma unroll
    for (int r = 0; r < 8; ++r) {
      float y = fmaf(acc[t][r], sc, sh);
      y = fmaxf(y, 0.0f);
      const int mo = (t << 4) + r + (hi << 3);
      const int wq = w0 + mo;
      if (mode == 0)
        outb[((long)(b * HPAD + h + 1) * WPAD + (wq + 1)) * 128 + co] = f2bf(y);
      else if (mode == 1)
        outb[((long)(pix0 + mo)) * 128 + co] = f2bf(y);
      else
        outf[(((long)b * 128 + co) * HH + h) * WW + wq] = y;
    }
  }
}

// ---------------------------------------------------------------------------
// linear GEMM (K = 128): out[pix][n] = A[pix][:128] . W[n][:128] + bias[n]
// Each wave: 2 M-tiles x 1 N-tile; K fully unrolled (4 chunks).
// mode 0: bf16 padded NHWC (128 ch); mode 1: f32 flat [pix][nstride]
// ---------------------------------------------------------------------------
__global__ __launch_bounds__(256) void lin_gemm(
    const unsigned short* __restrict__ A,      // [NPIX][128] bf16
    const unsigned short* __restrict__ wp,     // [Npad][128] bf16
    int Ntiles,
    const float* __restrict__ bias, int nbias,
    unsigned short* __restrict__ outb, float* __restrict__ outf,
    int mode, int nstride) {
  const int lane = threadIdx.x & 31;
  const int wv   = threadIdx.x >> 5;
  const int hi   = lane >> 4;
  const int lm   = lane & 15;
  const int ntile = blockIdx.y * 8 + wv;
  if (ntile >= Ntiles) return;             // wave-uniform
  const int pix0 = blockIdx.x << 5;

  const unsigned short* ap0 = A + (long)(pix0 + lm) * 128 + (hi << 3);
  const unsigned short* bp0 = wp + (long)((ntile << 4) + lm) * 128 + (hi << 3);

  v8f acc[2] = {{}, {}};
#pragma unroll
  for (int kk = 0; kk < 128; kk += 32) {
    BF16Frag a0, a1, bb;
    const unsigned short* ap = ap0 + kk;
    a0.q[0] = *(const u32x4*)(ap);
    a0.q[1] = *(const u32x4*)(ap + 16);
    a1.q[0] = *(const u32x4*)(ap + 16 * 128);
    a1.q[1] = *(const u32x4*)(ap + 16 * 128 + 16);
    const unsigned short* bq = bp0 + kk;
    bb.q[0] = *(const u32x4*)(bq);
    bb.q[1] = *(const u32x4*)(bq + 16);
    acc[0] = wmma_bf16(a0, bb, acc[0]);
    acc[1] = wmma_bf16(a1, bb, acc[1]);
  }

  const int n = (ntile << 4) + lm;
  const float bv = (n < nbias) ? bias[n] : 0.0f;
#pragma unroll
  for (int t = 0; t < 2; ++t) {
#pragma unroll
    for (int r = 0; r < 8; ++r) {
      float y = acc[t][r] + bv;
      const int pix = pix0 + (t << 4) + r + (hi << 3);
      if (mode == 0) {
        const int b = pix >> 14, hh = (pix >> 7) & 127, ww2 = pix & 127;
        outb[((long)(b * HPAD + hh + 1) * WPAD + (ww2 + 1)) * 128 + n] = f2bf(y);
      } else {
        outf[(long)pix * nstride + n] = y;
      }
    }
  }
}

// softmax over q (groups of 9), in place. one thread per (pixel, head, p)
__global__ void k_softmax9(float* __restrict__ a) {
  long t = (long)blockIdx.x * blockDim.x + threadIdx.x;
  if (t >= (long)NPIX * 36) return;
  long pix = t / 36;
  int hp = (int)(t % 36);
  float* row = a + pix * ATT_STRIDE + (hp / 9) * 81 + (hp % 9) * 9;
  float v[9], mx = -1e30f;
#pragma unroll
  for (int i = 0; i < 9; ++i) { v[i] = row[i] * ATT_SCALE; mx = fmaxf(mx, v[i]); }
  float s = 0.0f;
#pragma unroll
  for (int i = 0; i < 9; ++i) { v[i] = __expf(v[i] - mx); s += v[i]; }
  const float inv = 1.0f / s;
#pragma unroll
  for (int i = 0; i < 9; ++i) row[i] = v[i] * inv;
}

// fused einsum('bhlpq,bhlqd') + fold:
// out[pix,c] = sum_p sum_q attn[s(p)][head*81+p*9+q] * vpad[b][hs+qi][ws+qj][c]
__global__ __launch_bounds__(256) void k_apply_fold(
    const float* __restrict__ attn,            // [NPIX][336] f32 (softmaxed)
    const unsigned short* __restrict__ vpad,   // [NB][130][130][128] bf16
    unsigned short* __restrict__ outf) {       // [NPIX][128] bf16
  const int tid = threadIdx.x;
  const long pix = (long)blockIdx.x * 2 + (tid >> 7);
  const int c = tid & 127;
  const int head = c >> 5;
  const int b = (int)(pix >> 14), h = (int)((pix >> 7) & 127), w = (int)(pix & 127);
  float acc = 0.0f;
#pragma unroll
  for (int ki = 0; ki < 3; ++ki) {
    const int hs = h + 1 - ki;
    if ((unsigned)hs >= HH) continue;
#pragma unroll
    for (int kjj = 0; kjj < 3; ++kjj) {
      const int ws = w + 1 - kjj;
      if ((unsigned)ws >= WW) continue;
      const int p = ki * 3 + kjj;
      const long s = ((long)b << 14) + hs * WW + ws;
      const float* aw = attn + s * ATT_STRIDE + head * 81 + p * 9;
      const long vb = ((long)(b * HPAD + hs) * WPAD + ws) * 128 + c;
#pragma unroll
      for (int qi = 0; qi < 3; ++qi)
#pragma unroll
        for (int qj = 0; qj < 3; ++qj)
          acc = fmaf(aw[qi * 3 + qj], bf2f(vpad[vb + (qi * WPAD + qj) * 128]), acc);
    }
  }
  outf[pix * 128 + c] = f2bf(acc);
}

// ---------------------------------------------------------------------------
// workspace layout (bytes, all 256-aligned); total ~106 MB
// ---------------------------------------------------------------------------
#define SZ_XPAD    (2L*130*130*256*2)   // 17,305,600
#define SZ_PAD128  (2L*130*130*128*2)   //  8,652,800
#define SZ_FLAT128 (32768L*128*2)       //  8,388,608
#define SZ_LOGITS  (32768L*336*4)       // 44,040,192
#define O_XPAD    0L
#define O_APAD1   (O_XPAD   + SZ_XPAD)
#define O_VPAD    (O_APAD1  + SZ_PAD128)
#define O_XWPAD   (O_VPAD   + SZ_PAD128)
#define O_A2      (O_XWPAD  + SZ_PAD128)   // flat bf16; reused for fg/bg
#define O_FOLD    (O_A2     + SZ_FLAT128)
#define O_LOGITS  (O_FOLD   + SZ_FLAT128)
#define O_W1      (O_LOGITS + SZ_LOGITS)
#define O_W2      (O_W1     + 128L*2304*2)
#define O_WO1     (O_W2     + 128L*1152*2)
#define O_WO2     (O_WO1    + 128L*1152*2)
#define O_WV      (O_WO2    + 128L*1152*2)
#define O_WPROJ   (O_WV     + 128L*128*2)
#define O_WAFG    (O_WPROJ  + 128L*128*2)
#define O_WABG    (O_WAFG   + 336L*128*2)

extern "C" void kernel_launch(void* const* d_in, const int* in_sizes, int n_in,
                              void* d_out, int out_size, void* d_ws, size_t ws_size,
                              hipStream_t stream) {
  (void)in_sizes; (void)n_in; (void)out_size; (void)ws_size;
  // ---- inputs (setup_inputs dict order, params flattened in insertion order)
  const float* x      = (const float*)d_in[0];
  const float* fg     = (const float*)d_in[1];
  const float* bg     = (const float*)d_in[2];
  const float* in1_w  = (const float*)d_in[3];
  const float* in1_g  = (const float*)d_in[4];
  const float* in1_b  = (const float*)d_in[5];
  const float* in1_m  = (const float*)d_in[6];
  const float* in1_v  = (const float*)d_in[7];
  const float* in2_w  = (const float*)d_in[8];
  const float* in2_g  = (const float*)d_in[9];
  const float* in2_b  = (const float*)d_in[10];
  const float* in2_m  = (const float*)d_in[11];
  const float* in2_v  = (const float*)d_in[12];
  const float* v_w    = (const float*)d_in[13];
  const float* v_b    = (const float*)d_in[14];
  const float* afg_w  = (const float*)d_in[15];
  const float* afg_b  = (const float*)d_in[16];
  const float* abg_w  = (const float*)d_in[17];
  const float* abg_b  = (const float*)d_in[18];
  const float* proj_w = (const float*)d_in[19];
  const float* proj_b = (const float*)d_in[20];
  const float* o1_w   = (const float*)d_in[21];
  const float* o1_g   = (const float*)d_in[22];
  const float* o1_b   = (const float*)d_in[23];
  const float* o1_m   = (const float*)d_in[24];
  const float* o1_v   = (const float*)d_in[25];
  const float* o2_w   = (const float*)d_in[26];
  const float* o2_g   = (const float*)d_in[27];
  const float* o2_b   = (const float*)d_in[28];
  const float* o2_m   = (const float*)d_in[29];
  const float* o2_v   = (const float*)d_in[30];

  char* ws = (char*)d_ws;
  unsigned short* xpad  = (unsigned short*)(ws + O_XPAD);
  unsigned short* apad1 = (unsigned short*)(ws + O_APAD1);
  unsigned short* vpad  = (unsigned short*)(ws + O_VPAD);
  unsigned short* xwpad = (unsigned short*)(ws + O_XWPAD);
  unsigned short* a2    = (unsigned short*)(ws + O_A2);
  unsigned short* fold  = (unsigned short*)(ws + O_FOLD);
  float*          logit = (float*)(ws + O_LOGITS);
  unsigned short* w1    = (unsigned short*)(ws + O_W1);
  unsigned short* w2    = (unsigned short*)(ws + O_W2);
  unsigned short* wo1   = (unsigned short*)(ws + O_WO1);
  unsigned short* wo2   = (unsigned short*)(ws + O_WO2);
  unsigned short* wv    = (unsigned short*)(ws + O_WV);
  unsigned short* wproj = (unsigned short*)(ws + O_WPROJ);
  unsigned short* wafg  = (unsigned short*)(ws + O_WAFG);
  unsigned short* wabg  = (unsigned short*)(ws + O_WABG);
  float* out = (float*)d_out;

  const int T = 256;
  // zero all padded activation buffers (contiguous region xpad..xwpad)
  {
    long n4 = (SZ_XPAD + 3 * SZ_PAD128) / 16;
    k_fill0<<<(unsigned)((n4 + T - 1) / T), T, 0, stream>>>((u32x4*)xpad, n4);
  }
  // pack activations & weights
  k_nchw_to_padbf16<<<(2L*256*128*128)/T, T, 0, stream>>>(x, xpad, 256);
  k_pack_convw<<<(128L*9*256)/T, T, 0, stream>>>(in1_w, w1, 128, 256);
  k_pack_convw<<<(128L*9*128)/T, T, 0, stream>>>(in2_w, w2, 128, 128);
  k_pack_convw<<<(128L*9*128)/T, T, 0, stream>>>(o1_w, wo1, 128, 128);
  k_pack_convw<<<(128L*9*128)/T, T, 0, stream>>>(o2_w, wo2, 128, 128);
  k_pack_mat<<<(128L*128)/T, T, 0, stream>>>(v_w,    wv,    128, 128, 128);
  k_pack_mat<<<(128L*128)/T, T, 0, stream>>>(proj_w, wproj, 128, 128, 128);
  k_pack_mat<<<(336L*128)/T, T, 0, stream>>>(afg_w, wafg, 324, 128, 336);
  k_pack_mat<<<(336L*128)/T, T, 0, stream>>>(abg_w, wabg, 324, 128, 336);

  // conv1: x(256ch) -> apad1 (padded bf16)
  conv_gemm<256><<<MBLK, T, 0, stream>>>(xpad, w1, in1_g, in1_b, in1_m, in1_v,
                                         apad1, nullptr, 0);
  // conv2: apad1 -> a2 (flat bf16)
  conv_gemm<128><<<MBLK, T, 0, stream>>>(apad1, w2, in2_g, in2_b, in2_m, in2_v,
                                         a2, nullptr, 1);
  // v = a2 @ v_w^T + v_b  -> vpad
  lin_gemm<<<dim3(MBLK, 1), T, 0, stream>>>(a2, wv, 8, v_b, 128,
                                            vpad, nullptr, 0, 0);
  // ---- fg round
  k_nchw_to_flatbf16<<<(2L*128*128*128)/T, T, 0, stream>>>(fg, a2, 128);
  lin_gemm<<<dim3(MBLK, 3), T, 0, stream>>>(a2, wafg, 21, afg_b, ATT_N,
                                            nullptr, logit, 1, ATT_STRIDE);
  k_softmax9<<<(NPIX * 36L) / T, T, 0, stream>>>(logit);
  k_apply_fold<<<NPIX / 2, T, 0, stream>>>(logit, vpad, fold);
  lin_gemm<<<dim3(MBLK, 1), T, 0, stream>>>(fold, wproj, 8, proj_b, 128,
                                            xwpad, nullptr, 0, 0);
  // ---- bg round (v-source = xw_fg padded)
  k_nchw_to_flatbf16<<<(2L*128*128*128)/T, T, 0, stream>>>(bg, a2, 128);
  lin_gemm<<<dim3(MBLK, 3), T, 0, stream>>>(a2, wabg, 21, abg_b, ATT_N,
                                            nullptr, logit, 1, ATT_STRIDE);
  k_softmax9<<<(NPIX * 36L) / T, T, 0, stream>>>(logit);
  k_apply_fold<<<NPIX / 2, T, 0, stream>>>(logit, xwpad, fold);
  lin_gemm<<<dim3(MBLK, 1), T, 0, stream>>>(fold, wproj, 8, proj_b, 128,
                                            vpad, nullptr, 0, 0);   // reuse vpad
  // o1: vpad -> apad1 (padded bf16)
  conv_gemm<128><<<MBLK, T, 0, stream>>>(vpad, wo1, o1_g, o1_b, o1_m, o1_v,
                                         apad1, nullptr, 0);
  // o2: apad1 -> d_out (f32 NCHW)
  conv_gemm<128><<<MBLK, T, 0, stream>>>(apad1, wo2, o2_g, o2_b, o2_m, o2_v,
                                         nullptr, out, 2);
}
